// GraphConvWL_26560077758774
// MI455X (gfx1250) — matrix operand order repaired and checked by reading, not compiled, not run on hardware
//
#include <hip/hip_runtime.h>

#define N_NODES 100000
#define N_EDGES 1600000
#define D 128

typedef __attribute__((ext_vector_type(16))) _Float16 v16h;
typedef __attribute__((ext_vector_type(8)))  _Float16 v8h;
typedef __attribute__((ext_vector_type(8)))  float    v8f;
typedef __attribute__((ext_vector_type(4)))  float    v4f;

// ---------------------------------------------------------------------------
// Phase 0: zero the aggregation scratch (N_NODES x D floats in d_ws).
// ---------------------------------------------------------------------------
__global__ void GraphConv_zero(float* __restrict__ agg) {
    size_t i = (size_t)blockIdx.x * blockDim.x + threadIdx.x;   // one float4 each
    v4f z = {0.f, 0.f, 0.f, 0.f};
    ((v4f*)agg)[i] = z;
}

// ---------------------------------------------------------------------------
// Phase 1: edge scatter  agg[dst] += w * feat[src].
// One wave (32 lanes) per edge, 4 floats per lane, native f32 global atomics.
// feat and agg (51.2 MB each) are L2-resident (192 MB L2), so the atomics
// resolve in L2 at device scope.
// ---------------------------------------------------------------------------
__global__ __launch_bounds__(256) void GraphConv_scatter(
        const float* __restrict__ feat,
        const int*   __restrict__ src,
        const int*   __restrict__ dst,
        const float* __restrict__ ew,
        float*       __restrict__ agg) {
    const int lane = threadIdx.x & 31;
    const int e    = blockIdx.x * (blockDim.x >> 5) + (threadIdx.x >> 5);

    const int   s = src[e];
    const int   d = dst[e];
    const float w = ew[e];

    v4f f = ((const v4f*)(feat + (size_t)s * D))[lane];   // global_load_b128
    float* p = agg + (size_t)d * D + lane * 4;

    __hip_atomic_fetch_add(p + 0, f.x * w, __ATOMIC_RELAXED, __HIP_MEMORY_SCOPE_AGENT);
    __hip_atomic_fetch_add(p + 1, f.y * w, __ATOMIC_RELAXED, __HIP_MEMORY_SCOPE_AGENT);
    __hip_atomic_fetch_add(p + 2, f.z * w, __ATOMIC_RELAXED, __HIP_MEMORY_SCOPE_AGENT);
    __hip_atomic_fetch_add(p + 3, f.w * w, __ATOMIC_RELAXED, __HIP_MEMORY_SCOPE_AGENT);
}

// ---------------------------------------------------------------------------
// Phase 2: fused dual GEMM  out = agg @ Wn + bn + feat @ Ws
// v_wmma_f32_16x16x32_f16, f16 operands, f32 accumulation.
// Block = 256 threads = 8 waves; wave w owns output columns [16w, 16w+16);
// block owns rows [16*blockIdx.x, +16).  K = 128 -> 4 k-steps x 2 WMMAs.
// Weights staged transposed (n-major) as f16 in LDS (2 x 32 KB).
// ---------------------------------------------------------------------------
__global__ __launch_bounds__(256) void GraphConv_gemm(
        const float* __restrict__ agg,
        const float* __restrict__ feat,
        const float* __restrict__ Wn,
        const float* __restrict__ Ws,
        const float* __restrict__ bn,
        float*       __restrict__ out) {
    __shared__ _Float16 sWn[D * D];   // sWn[n*D + k] = (f16)Wn[k*D + n]
    __shared__ _Float16 sWs[D * D];

    for (int i = threadIdx.x; i < D * D; i += 256) {
        int k = i >> 7;
        int n = i & (D - 1);
        sWn[n * D + k] = (_Float16)Wn[i];
        sWs[n * D + k] = (_Float16)Ws[i];
    }
    __syncthreads();

    const int lane  = threadIdx.x & 31;
    const int wave  = threadIdx.x >> 5;          // column tile 0..7
    const int row0  = blockIdx.x * 16;
    const int nloc  = lane & 15;
    const int col   = wave * 16 + nloc;
    const int kgA   = (lane >> 4) << 3;          // A: K subgroup 0 or 8
    const int kgB   = (lane >> 4) << 3;          // B: K subgroup 0 or 8
    const int arow  = row0 + nloc;               // A row for this lane (M = lane%16)

    const v4f* agg4  = (const v4f*)(agg  + (size_t)arow * D);
    const v4f* feat4 = (const v4f*)(feat + (size_t)arow * D);

    v8f acc = {};

#pragma unroll
    for (int kb = 0; kb < D; kb += 32) {
        // ---- A fragments (16x32 f16 layout): lane holds K = kgA..kgA+7 and
        //      K = 16+kgA..16+kgA+7 of row (lane%16), relative to kb. ----
        const int q = (kb + kgA) >> 2;           // float4 index
        v4f a0 = agg4[q],  a1 = agg4[q + 1],  a2 = agg4[q + 4],  a3 = agg4[q + 5];
        v4f f0 = feat4[q], f1 = feat4[q + 1], f2 = feat4[q + 4], f3 = feat4[q + 5];

        union { v16h v; _Float16 h[16]; } Aa, Af;
        Aa.h[0]  = (_Float16)a0.x; Aa.h[1]  = (_Float16)a0.y;
        Aa.h[2]  = (_Float16)a0.z; Aa.h[3]  = (_Float16)a0.w;
        Aa.h[4]  = (_Float16)a1.x; Aa.h[5]  = (_Float16)a1.y;
        Aa.h[6]  = (_Float16)a1.z; Aa.h[7]  = (_Float16)a1.w;
        Aa.h[8]  = (_Float16)a2.x; Aa.h[9]  = (_Float16)a2.y;
        Aa.h[10] = (_Float16)a2.z; Aa.h[11] = (_Float16)a2.w;
        Aa.h[12] = (_Float16)a3.x; Aa.h[13] = (_Float16)a3.y;
        Aa.h[14] = (_Float16)a3.z; Aa.h[15] = (_Float16)a3.w;

        Af.h[0]  = (_Float16)f0.x; Af.h[1]  = (_Float16)f0.y;
        Af.h[2]  = (_Float16)f0.z; Af.h[3]  = (_Float16)f0.w;
        Af.h[4]  = (_Float16)f1.x; Af.h[5]  = (_Float16)f1.y;
        Af.h[6]  = (_Float16)f1.z; Af.h[7]  = (_Float16)f1.w;
        Af.h[8]  = (_Float16)f2.x; Af.h[9]  = (_Float16)f2.y;
        Af.h[10] = (_Float16)f2.z; Af.h[11] = (_Float16)f2.w;
        Af.h[12] = (_Float16)f3.x; Af.h[13] = (_Float16)f3.y;
        Af.h[14] = (_Float16)f3.z; Af.h[15] = (_Float16)f3.w;

        // ---- B fragments (32x16 f16 layout): lane = column N, halves are
        //      contiguous in K thanks to transposed LDS staging: two 16B reads.
        union { v16h v; v8h q2[2]; } Bn_, Bs_;
        const _Float16* cn = sWn + (size_t)col * D + kb + kgB;
        const _Float16* cs = sWs + (size_t)col * D + kb + kgB;
        Bn_.q2[0] = *(const v8h*)(cn);
        Bn_.q2[1] = *(const v8h*)(cn + 16);
        Bs_.q2[0] = *(const v8h*)(cs);
        Bs_.q2[1] = *(const v8h*)(cs + 16);

        acc = __builtin_amdgcn_wmma_f32_16x16x32_f16(
                  false, Aa.v, false, Bn_.v, (short)0, acc, false, false);
        acc = __builtin_amdgcn_wmma_f32_16x16x32_f16(
                  false, Af.v, false, Bs_.v, (short)0, acc, false, false);
    }

    // ---- Epilogue: C/D layout — lane%16 = N, VGPR r = row r (+8 for lanes>=16).
    const float bias = bn[col];
    const int   m0   = (lane >> 4) << 3;
#pragma unroll
    for (int r = 0; r < 8; ++r) {
        out[(size_t)(row0 + m0 + r) * D + col] = acc[r] + bias;
    }
}

// ---------------------------------------------------------------------------
extern "C" void kernel_launch(void* const* d_in, const int* in_sizes, int n_in,
                              void* d_out, int out_size, void* d_ws, size_t ws_size,
                              hipStream_t stream) {
    const float* feat = (const float*)d_in[0];
    const int*   src  = (const int*)  d_in[1];
    const int*   dst  = (const int*)  d_in[2];
    const float* ew   = (const float*)d_in[3];
    const float* Wn   = (const float*)d_in[4];
    const float* Ws   = (const float*)d_in[5];
    const float* bn   = (const float*)d_in[6];
    float*       out  = (float*)d_out;
    float*       agg  = (float*)d_ws;            // N_NODES * D floats (51.2 MB)

    // Phase 0: zero agg (3.2M float4 stores).
    GraphConv_zero<<<(N_NODES * D / 4) / 256, 256, 0, stream>>>(agg);

    // Phase 1: edge scatter, 8 edges per 256-thread block.
    GraphConv_scatter<<<N_EDGES / 8, 256, 0, stream>>>(feat, src, dst, ew, agg);

    // Phase 2: fused WMMA GEMM, 16-row panels.
    GraphConv_gemm<<<N_NODES / 16, 256, 0, stream>>>(agg, feat, Wn, Ws, bn, out);
}
